// SyntheticGRU_7078106104378
// MI455X (gfx1250) — compile-verified, matrix operand order checked by reading, not run on hardware
//
#include <hip/hip_runtime.h>
#include <hip/hip_bf16.h>

typedef __attribute__((ext_vector_type(16))) __bf16 v16bf;
typedef __attribute__((ext_vector_type(8)))  __bf16 v8bf;
typedef __attribute__((ext_vector_type(8)))  float  v8f;

#define CC  4096   // concepts
#define HH  1024   // hidden
#define TT  32     // time steps
#define BB  512    // batch
#define G3H 3072   // 3*H

#if __has_builtin(__builtin_amdgcn_global_load_async_to_lds_b128) && \
    __has_builtin(__builtin_amdgcn_s_wait_asynccnt)
#define HAVE_ASYNC_LDS 1
#else
#define HAVE_ASYNC_LDS 0
#endif

#if HAVE_ASYNC_LDS
// The builtin's pointer params are GCC-vector int4 with explicit address spaces:
//   arg0: v4i addrspace(1)*  (global source)
//   arg1: v4i addrspace(3)*  (LDS destination)
typedef int v4i_g __attribute__((vector_size(16)));
typedef __attribute__((address_space(1))) v4i_g* glb_v4i;
typedef __attribute__((address_space(3))) v4i_g* lds_v4i;
#endif

// ---------------------------------------------------------------------------
// fp32 -> bf16 conversion (grid-stride)
// ---------------------------------------------------------------------------
__global__ void f32_to_bf16_kernel(const float* __restrict__ in,
                                   __bf16* __restrict__ out, int n) {
    int stride = gridDim.x * blockDim.x;
    for (int i = blockIdx.x * blockDim.x + threadIdx.x; i < n; i += stride)
        out[i] = (__bf16)in[i];
}

__global__ void zero_h_kernel(float* __restrict__ hf, __bf16* __restrict__ hb, int n) {
    int i = blockIdx.x * blockDim.x + threadIdx.x;
    if (i < n) { hf[i] = 0.0f; hb[i] = (__bf16)0.0f; }
}

// ---------------------------------------------------------------------------
// WMMA GEMM:  D(MxN) = A(MxK) @ W(NxK)^T  [+ bias, sigmoid]   (bf16 in, f32 acc)
// Block tile 128(M) x 256(N), 256 threads = 8 waves, wave tile 64x64
// (4x4 grid of 16x16 f32 accumulators). K staged through double-buffered LDS
// in 32-wide slabs, filled by GLOBAL_LOAD_ASYNC_TO_LDS_B128 (ASYNCcnt) when
// available so no staging VGPRs are live across the WMMA burst.
// ---------------------------------------------------------------------------
template <bool SIG>
__global__ __launch_bounds__(256, 1)
void wmma_gemm_kernel(const __bf16* __restrict__ A,
                      const __bf16* __restrict__ W,
                      const float*  __restrict__ bias,
                      float*        __restrict__ outF, long outStride,
                      __bf16*       __restrict__ outB,
                      int M, int N, int K) {
    __shared__ __bf16 As [2][128 * 40];   // 2 x 10 KB (stride 40 halves, 16B aligned)
    __shared__ __bf16 Wsm[2][256 * 40];   // 2 x 20 KB

    const int tid  = threadIdx.x;
    const int lane = tid & 31;
    const int wave = tid >> 5;
    const int wm   = wave & 1;    // 0..1  -> 64-row half of M-tile
    const int wn   = wave >> 1;   // 0..3  -> 64-col quarter of N-tile
    const int bm   = blockIdx.y * 128;
    const int bn   = blockIdx.x * 256;

    v8f acc[4][4];
#pragma unroll
    for (int mi = 0; mi < 4; ++mi)
#pragma unroll
        for (int ni = 0; ni < 4; ++ni)
#pragma unroll
            for (int e = 0; e < 8; ++e) acc[mi][ni][e] = 0.0f;

    const int fr = lane & 15;          // fragment row within 16-tile
    const int kh = (lane >> 4) * 8;    // K half-select: 0 or 8
    const int nIter = K >> 5;

#if HAVE_ASYNC_LDS
    // ---- async DMA staging: global -> LDS, no VGPR round trip ----
    auto asyncSlab = [&](int kk, int buf) {
#pragma unroll
        for (int j = 0; j < 2; ++j) {
            int c = tid + 256 * j; int row = c >> 2; int cc = (c & 3) * 8;
            v4i_g* gp = (v4i_g*)(A + (size_t)(bm + row) * K + kk + cc);
            v4i_g* lp = (v4i_g*)(&As[buf][row * 40 + cc]);
            __builtin_amdgcn_global_load_async_to_lds_b128(
                (glb_v4i)gp, (lds_v4i)lp, 0, 0);
        }
#pragma unroll
        for (int j = 0; j < 4; ++j) {
            int c = tid + 256 * j; int row = c >> 2; int cc = (c & 3) * 8;
            v4i_g* gp = (v4i_g*)(W + (size_t)(bn + row) * K + kk + cc);
            v4i_g* lp = (v4i_g*)(&Wsm[buf][row * 40 + cc]);
            __builtin_amdgcn_global_load_async_to_lds_b128(
                (glb_v4i)gp, (lds_v4i)lp, 0, 0);
        }
    };
    asyncSlab(0, 0);
    __builtin_amdgcn_s_wait_asynccnt(0);
    __syncthreads();
#else
    // ---- fallback: register staging ----
    uint4 aR[2], wR[4];
    auto loadSlab = [&](int kk) {
#pragma unroll
        for (int j = 0; j < 2; ++j) {
            int c = tid + 256 * j; int row = c >> 2; int cc = (c & 3) * 8;
            aR[j] = *(const uint4*)(A + (size_t)(bm + row) * K + kk + cc);
        }
#pragma unroll
        for (int j = 0; j < 4; ++j) {
            int c = tid + 256 * j; int row = c >> 2; int cc = (c & 3) * 8;
            wR[j] = *(const uint4*)(W + (size_t)(bn + row) * K + kk + cc);
        }
    };
    auto storeSlab = [&](int buf) {
#pragma unroll
        for (int j = 0; j < 2; ++j) {
            int c = tid + 256 * j; int row = c >> 2; int cc = (c & 3) * 8;
            *(uint4*)(&As[buf][row * 40 + cc]) = aR[j];
        }
#pragma unroll
        for (int j = 0; j < 4; ++j) {
            int c = tid + 256 * j; int row = c >> 2; int cc = (c & 3) * 8;
            *(uint4*)(&Wsm[buf][row * 40 + cc]) = wR[j];
        }
    };
    loadSlab(0);
    storeSlab(0);
    __syncthreads();
#endif

    for (int it = 0; it < nIter; ++it) {
        const int buf = it & 1;
#if HAVE_ASYNC_LDS
        if (it + 1 < nIter) asyncSlab((it + 1) << 5, buf ^ 1);   // fill other buffer
#else
        if (it + 1 < nIter) loadSlab((it + 1) << 5);             // prefetch to regs
#endif

        // ---- fragments from current LDS buffer ----
        v16bf afrag[4], wfrag[4];
#pragma unroll
        for (int mi = 0; mi < 4; ++mi) {
            int row = wm * 64 + mi * 16 + fr;
            v8bf lo = *(const v8bf*)(&As[buf][row * 40 + kh]);
            v8bf hi = *(const v8bf*)(&As[buf][row * 40 + kh + 16]);
#pragma unroll
            for (int e = 0; e < 8; ++e) { afrag[mi][e] = lo[e]; afrag[mi][e + 8] = hi[e]; }
        }
#pragma unroll
        for (int ni = 0; ni < 4; ++ni) {
            int row = wn * 64 + ni * 16 + fr;
            v8bf lo = *(const v8bf*)(&Wsm[buf][row * 40 + kh]);
            v8bf hi = *(const v8bf*)(&Wsm[buf][row * 40 + kh + 16]);
#pragma unroll
            for (int e = 0; e < 8; ++e) { wfrag[ni][e] = lo[e]; wfrag[ni][e + 8] = hi[e]; }
        }

        // ---- 16 WMMAs per K-slab per wave ----
#pragma unroll
        for (int mi = 0; mi < 4; ++mi)
#pragma unroll
            for (int ni = 0; ni < 4; ++ni)
                acc[mi][ni] = __builtin_amdgcn_wmma_f32_16x16x32_bf16(
                    false, afrag[mi], false, wfrag[ni],
                    (short)0, acc[mi][ni], false, false);

        if (it + 1 < nIter) {
#if HAVE_ASYNC_LDS
            __builtin_amdgcn_s_wait_asynccnt(0);
            __syncthreads();
#else
            storeSlab(buf ^ 1);
            __syncthreads();
#endif
        }
    }

    // ---- epilogue: C/D layout -> lane holds N=lane&15, VGPR r holds M=r(+8) ----
    const int fn   = lane & 15;
    const int moff = (lane >> 4) * 8;
#pragma unroll
    for (int mi = 0; mi < 4; ++mi) {
#pragma unroll
        for (int ni = 0; ni < 4; ++ni) {
            int nglob = bn + wn * 64 + ni * 16 + fn;
            float bv = SIG ? bias[nglob] : 0.0f;
#pragma unroll
            for (int r = 0; r < 8; ++r) {
                int mglob = bm + wm * 64 + mi * 16 + moff + r;
                float v = acc[mi][ni][r] + bv;
                if (SIG) {
                    v = 1.0f / (1.0f + __expf(-v));
                    outF[(size_t)mglob * outStride + nglob] = v;
                    outB[(size_t)mglob * N + nglob] = (__bf16)v;
                } else {
                    outF[(size_t)mglob * outStride + nglob] = v;
                }
            }
        }
    }
}

// ---------------------------------------------------------------------------
// GRU gate fusion: r,z = sigmoid, n = tanh(i_n + r*h_n), h' = (1-z)n + z h
// ---------------------------------------------------------------------------
__global__ void gru_fuse_kernel(const float* __restrict__ gi,
                                const float* __restrict__ gh,
                                const float* __restrict__ b_ih,
                                const float* __restrict__ b_hh,
                                const float* __restrict__ h_prev,
                                float*  __restrict__ h_ws,
                                __bf16* __restrict__ h_bf,
                                float*  __restrict__ h_out, long hidStride) {
    int idx = blockIdx.x * blockDim.x + threadIdx.x;   // 0 .. B*H-1
    int b = idx / HH, j = idx % HH;
    size_t base = (size_t)b * G3H;
    float ir  = gi[base + j]           + b_ih[j];
    float iz  = gi[base + HH + j]      + b_ih[HH + j];
    float in_ = gi[base + 2*HH + j]    + b_ih[2*HH + j];
    float hr  = gh[base + j]           + b_hh[j];
    float hz  = gh[base + HH + j]      + b_hh[HH + j];
    float hn  = gh[base + 2*HH + j]    + b_hh[2*HH + j];
    float r = 1.0f / (1.0f + __expf(-(ir + hr)));
    float z = 1.0f / (1.0f + __expf(-(iz + hz)));
    float n = tanhf(in_ + r * hn);
    float hp = h_prev[idx];
    float hnew = (1.0f - z) * n + z * hp;
    h_ws[idx] = hnew;
    h_bf[idx] = (__bf16)hnew;
    h_out[(size_t)b * hidStride + j] = hnew;
}

// ---------------------------------------------------------------------------
extern "C" void kernel_launch(void* const* d_in, const int* in_sizes, int n_in,
                              void* d_out, int out_size, void* d_ws, size_t ws_size,
                              hipStream_t stream) {
    const float* noise = (const float*)d_in[0];   // (B,H)
    const float* W_ih  = (const float*)d_in[1];   // (3H,C)
    const float* W_hh  = (const float*)d_in[2];   // (3H,H)
    const float* b_ih  = (const float*)d_in[3];   // (3H,)
    const float* b_hh  = (const float*)d_in[4];   // (3H,)
    const float* W_lin = (const float*)d_in[5];   // (C,H)
    const float* b_lin = (const float*)d_in[6];   // (C,)

    float* probs = (float*)d_out;                         // (B,T,C)
    float* hids  = (float*)d_out + (size_t)BB * TT * CC;  // (B,T,H)

    char* ws = (char*)d_ws;
    auto carve = [&](size_t bytes) -> char* {
        char* p = ws; ws += (bytes + 255) & ~(size_t)255; return p;
    };
    __bf16* Wih_b   = (__bf16*)carve((size_t)G3H * CC * 2);
    __bf16* Whh_b   = (__bf16*)carve((size_t)G3H * HH * 2);
    __bf16* Wlin_b  = (__bf16*)carve((size_t)CC  * HH * 2);
    __bf16* noise_b = (__bf16*)carve((size_t)BB  * HH * 2);
    __bf16* p_b     = (__bf16*)carve((size_t)BB  * CC * 2);
    __bf16* h_b     = (__bf16*)carve((size_t)BB  * HH * 2);
    float*  h_f     = (float*) carve((size_t)BB  * HH * 4);
    float*  gi      = (float*) carve((size_t)BB  * G3H * 4);
    float*  gh      = (float*) carve((size_t)BB  * G3H * 4);

    // --- weight / input conversion to bf16 ---
    f32_to_bf16_kernel<<<2048, 256, 0, stream>>>(W_ih,  Wih_b,  G3H * CC);
    f32_to_bf16_kernel<<<1024, 256, 0, stream>>>(W_hh,  Whh_b,  G3H * HH);
    f32_to_bf16_kernel<<<1024, 256, 0, stream>>>(W_lin, Wlin_b, CC * HH);
    f32_to_bf16_kernel<<<256,  256, 0, stream>>>(noise, noise_b, BB * HH);
    zero_h_kernel<<<(BB * HH) / 256, 256, 0, stream>>>(h_f, h_b, BB * HH);

    dim3 gP(CC  / 256, BB / 128);   // logits GEMM grid (16,4)
    dim3 gG(G3H / 256, BB / 128);   // gates  GEMM grid (12,4)

    // p0 = sigmoid(noise @ W_lin^T + b_lin) -> probs[:,0,:]
    wmma_gemm_kernel<true><<<gP, 256, 0, stream>>>(
        noise_b, Wlin_b, b_lin, probs, (long)TT * CC, p_b, BB, CC, HH);

    for (int t = 0; t < TT; ++t) {
        // gi = p @ W_ih^T     (K = C)
        wmma_gemm_kernel<false><<<gG, 256, 0, stream>>>(
            p_b, Wih_b, nullptr, gi, (long)G3H, nullptr, BB, G3H, CC);
        // gh = h @ W_hh^T     (K = H)
        wmma_gemm_kernel<false><<<gG, 256, 0, stream>>>(
            h_b, Whh_b, nullptr, gh, (long)G3H, nullptr, BB, G3H, HH);
        // h_{t+1} = GRU(gi, gh, h_t)  -> hiddens[:,t,:]
        gru_fuse_kernel<<<(BB * HH) / 256, 256, 0, stream>>>(
            gi, gh, b_ih, b_hh, h_f, h_f, h_b, hids + (size_t)t * HH, (long)TT * HH);
        // p_{t+1} = sigmoid(h_{t+1} @ W_lin^T + b_lin) -> probs[:,t+1,:]
        if (t + 1 < TT)
            wmma_gemm_kernel<true><<<gP, 256, 0, stream>>>(
                h_b, Wlin_b, b_lin, probs + (size_t)(t + 1) * CC, (long)TT * CC,
                p_b, BB, CC, HH);
    }
}
